// Mamba_64656437674699
// MI455X (gfx1250) — compile-verified, matrix operand order checked by reading, not run on hardware
//
#include <hip/hip_runtime.h>
#include <hip/hip_bf16.h>
#include <math.h>

// ---------------------------------------------------------------------------
// Mamba forward for MI455X (gfx1250, wave32, WMMA).
// B=8 L=256 D_MODEL=1024 D_INNER=2048 N_STATE=16 DT_RANK=64 D_CONV=4 N_LAYER=2
// ---------------------------------------------------------------------------

#define BB      8
#define LL      256
#define MROWS   (BB * LL)        // 2048
#define DM      1024
#define DI      2048
#define NS      16
#define DR      64
#define NCLS    1000
#define EPSV    1e-5f

typedef __attribute__((ext_vector_type(16))) __bf16 v16bf;
typedef __attribute__((ext_vector_type(8)))  float  v8f;

// ---------------------------------------------------------------------------
// Embedding gather: x[row][c] = emb[ids[row]][c]
// ---------------------------------------------------------------------------
__global__ void embed_kernel(const int* __restrict__ ids,
                             const float* __restrict__ emb,
                             float* __restrict__ x, int total) {
  int i = blockIdx.x * blockDim.x + threadIdx.x;
  if (i >= total) return;
  int row = i / DM, c = i % DM;
  x[i] = emb[(size_t)ids[row] * DM + c];
}

// ---------------------------------------------------------------------------
// RMSNorm over D columns, one block per row, writes bf16 (GEMM input)
// ---------------------------------------------------------------------------
__global__ void rmsnorm_bf16_kernel(const float* __restrict__ x,
                                    const float* __restrict__ w,
                                    __bf16* __restrict__ out, int D) {
  __shared__ float red[256];
  int row = blockIdx.x;
  const float* xr = x + (size_t)row * D;
  float s = 0.f;
  for (int c = threadIdx.x; c < D; c += blockDim.x) { float v = xr[c]; s += v * v; }
  red[threadIdx.x] = s; __syncthreads();
  for (int st = 128; st > 0; st >>= 1) {
    if ((int)threadIdx.x < st) red[threadIdx.x] += red[threadIdx.x + st];
    __syncthreads();
  }
  float scale = rsqrtf(red[0] / (float)D + EPSV);
  for (int c = threadIdx.x; c < D; c += blockDim.x)
    out[(size_t)row * D + c] = (__bf16)(xr[c] * scale * w[c]);
}

// Final RMSNorm: only the last token of each sequence, fp32 out (8 x 1024)
__global__ void rmsnorm_last_kernel(const float* __restrict__ x,
                                    const float* __restrict__ w,
                                    float* __restrict__ out) {
  __shared__ float red[256];
  int b = blockIdx.x;
  int row = b * LL + (LL - 1);
  const float* xr = x + (size_t)row * DM;
  float s = 0.f;
  for (int c = threadIdx.x; c < DM; c += blockDim.x) { float v = xr[c]; s += v * v; }
  red[threadIdx.x] = s; __syncthreads();
  for (int st = 128; st > 0; st >>= 1) {
    if ((int)threadIdx.x < st) red[threadIdx.x] += red[threadIdx.x + st];
    __syncthreads();
  }
  float scale = rsqrtf(red[0] / (float)DM + EPSV);
  for (int c = threadIdx.x; c < DM; c += blockDim.x)
    out[(size_t)b * DM + c] = xr[c] * scale * w[c];
}

// ---------------------------------------------------------------------------
// Weight convert: W[K][N] fp32  ->  WT[N][K] bf16 (so B-fragments are contig)
// ---------------------------------------------------------------------------
__global__ void transpose_bf16_kernel(const float* __restrict__ W,
                                      __bf16* __restrict__ WT, int K, int N) {
  int i = blockIdx.x * blockDim.x + threadIdx.x;
  if (i >= K * N) return;
  int k = i / N, n = i % N;
  WT[(size_t)n * K + k] = (__bf16)W[i];
}

// ---------------------------------------------------------------------------
// WMMA GEMM: C[M][N] = A[M][K](bf16) x WT[N][K](bf16), f32 accumulate.
// Fully compile-time tiling: each wave owns a 16(M) x NPW(N) strip with
// NPW/16 accumulators (A fragment reused). N % NPW == 0, K % 32 == 0 and
// the grid divides exactly, so there are NO runtime guards -> no EXEC
// manipulation around v_wmma (ISA requires EXEC all-ones).
// ---------------------------------------------------------------------------
template <int M, int N, int K, int NPW>
__global__ void gemm_bf16_wmma_kernel(const __bf16* __restrict__ A,
                                      const __bf16* __restrict__ WT,
                                      float* __restrict__ C) {
  static_assert(N % NPW == 0 && K % 32 == 0 && M % 16 == 0, "tile mismatch");
  constexpr int TILES = NPW / 16;
  constexpr int NSTRIPS = N / NPW;

  int wave = threadIdx.x >> 5;
  int lane = threadIdx.x & 31;
  int gw   = blockIdx.x * (blockDim.x >> 5) + wave;
  int m0 = (gw / NSTRIPS) * 16;
  int n0 = (gw % NSTRIPS) * NPW;

  int l15   = lane & 15;
  int khalf = (lane >> 4) * 16;            // which 16 of the K=32 slab

  const __bf16* aptr = A  + (size_t)(m0 + l15) * K + khalf;
  const __bf16* bptr = WT + (size_t)(n0 + l15) * K + khalf;

  v8f acc[TILES];
#pragma unroll
  for (int t = 0; t < TILES; ++t) acc[t] = (v8f){};

  for (int kb = 0; kb < K; kb += 32) {
    v16bf a = *(const v16bf*)(aptr + kb);
#pragma unroll
    for (int t = 0; t < TILES; ++t) {
      v16bf b = *(const v16bf*)(bptr + (size_t)t * 16 * K + kb);
      acc[t] = __builtin_amdgcn_wmma_f32_16x16x32_bf16(false, a, false, b,
                                                       (short)0, acc[t],
                                                       false, false);
    }
  }

  // C/D layout: element r -> row m0 + r + 8*(lane>>4), col = tile_n + lane&15
  int rbase = m0 + ((lane >> 4) << 3);
#pragma unroll
  for (int t = 0; t < TILES; ++t) {
    int col = n0 + t * 16 + l15;
#pragma unroll
    for (int r = 0; r < 8; ++r)
      C[(size_t)(rbase + r) * N + col] = acc[t][r];
  }
}

// ---------------------------------------------------------------------------
// Depthwise causal conv (width 4) + bias on xi = xr[:, :DI]; writes f32+bf16
// ---------------------------------------------------------------------------
__global__ void conv_kernel(const float* __restrict__ xr,
                            const float* __restrict__ w,
                            const float* __restrict__ bias,
                            float* __restrict__ xc,
                            __bf16* __restrict__ xcb, int total) {
  int i = blockIdx.x * blockDim.x + threadIdx.x;
  if (i >= total) return;
  int row = i / DI, d = i % DI;
  int t = row % LL;
  float acc = bias[d];
#pragma unroll
  for (int k = 0; k < 4; ++k) {
    int dt = k - 3;                      // source offset (causal pad = 3)
    if (t + dt >= 0)
      acc += w[d * 4 + k] * xr[(size_t)(row + dt) * (2 * DI) + d];
  }
  xc[i] = acc;
  xcb[i] = (__bf16)acc;
}

// strided fp32 -> contiguous bf16 for delta = x_dbl[:, :DR]
__global__ void delta_bf16_kernel(const float* __restrict__ xdbl,
                                  __bf16* __restrict__ out, int total) {
  int i = blockIdx.x * blockDim.x + threadIdx.x;
  if (i >= total) return;
  int row = i / DR, c = i % DR;
  out[i] = (__bf16)xdbl[(size_t)row * (DR + 2 * NS) + c];
}

// delta = softplus(dtp + b), in place
__global__ void softplus_kernel(float* __restrict__ dtp,
                                const float* __restrict__ b, int total) {
  int i = blockIdx.x * blockDim.x + threadIdx.x;
  if (i >= total) return;
  int d = i % DI;
  float v = dtp[i] + b[d];
  dtp[i] = (v > 20.f) ? v : log1pf(expf(v));
}

// ---------------------------------------------------------------------------
// Selective scan: one thread per (b, d), 16 states in registers, sequential
// over L.  deltaA / dBu recomputed on the fly (saves 536MB of intermediates).
// ---------------------------------------------------------------------------
__global__ void scan_kernel(const float* __restrict__ dt,
                            const float* __restrict__ xc,
                            const float* __restrict__ xdbl,
                            const float* __restrict__ A_log,
                            float* __restrict__ y) {
  int idx = blockIdx.x * blockDim.x + threadIdx.x;
  if (idx >= BB * DI) return;
  int b = idx / DI, d = idx % DI;

  float Av[NS], h[NS];
#pragma unroll
  for (int n = 0; n < NS; ++n) { Av[n] = -expf(A_log[d * NS + n]); h[n] = 0.f; }

  for (int t = 0; t < LL; ++t) {
    size_t row = (size_t)b * LL + t;
    float dv = dt[row * DI + d];
    float xv = xc[row * DI + d];
    float du = dv * xv;
    const float* bc = xdbl + row * (DR + 2 * NS) + DR;   // Bm | C
    float acc = 0.f;
#pragma unroll
    for (int n = 0; n < NS; ++n) {
      h[n] = expf(dv * Av[n]) * h[n] + du * bc[n];
      acc += h[n] * bc[NS + n];
    }
    y[row * DI + d] = acc;
  }
}

// y = (yscan + xi*D) * silu(res); writes bf16 for out_proj
__global__ void gate_kernel(const float* __restrict__ ysc,
                            const float* __restrict__ xc,
                            const float* __restrict__ Dp,
                            const float* __restrict__ xr,
                            __bf16* __restrict__ ybf, int total) {
  int i = blockIdx.x * blockDim.x + threadIdx.x;
  if (i >= total) return;
  int row = i / DI, d = i % DI;
  float yv = ysc[i] + xc[i] * Dp[d];
  float r = xr[(size_t)row * (2 * DI) + DI + d];
  yv *= r / (1.f + expf(-r));
  ybf[i] = (__bf16)yv;
}

__global__ void add_kernel(float* __restrict__ x,
                           const float* __restrict__ t, int total) {
  int i = blockIdx.x * blockDim.x + threadIdx.x;
  if (i < total) x[i] += t[i];
}

// logits[b][c] = xf[b] . fc_w[:, c] + fc_b[c]   (8x1024x1000, tiny)
__global__ void fc_kernel(const float* __restrict__ xf,
                          const float* __restrict__ fcw,
                          const float* __restrict__ fcb,
                          float* __restrict__ out, int total) {
  int i = blockIdx.x * blockDim.x + threadIdx.x;
  if (i >= total) return;
  int b = i / NCLS, c = i % NCLS;
  float s = fcb[c];
  for (int k = 0; k < DM; ++k)
    s += xf[(size_t)b * DM + k] * fcw[(size_t)k * NCLS + c];
  out[i] = s;
}

// softmax over 1000 classes; one block per row; out at offset 8*1000
__global__ void softmax_kernel(float* __restrict__ out) {
  __shared__ float red[256];
  int b = blockIdx.x;
  const float* lg = out + (size_t)b * NCLS;
  float m = -1e30f;
  for (int c = threadIdx.x; c < NCLS; c += blockDim.x) m = fmaxf(m, lg[c]);
  red[threadIdx.x] = m; __syncthreads();
  for (int st = 128; st > 0; st >>= 1) {
    if ((int)threadIdx.x < st)
      red[threadIdx.x] = fmaxf(red[threadIdx.x], red[threadIdx.x + st]);
    __syncthreads();
  }
  m = red[0]; __syncthreads();
  float s = 0.f;
  for (int c = threadIdx.x; c < NCLS; c += blockDim.x) s += expf(lg[c] - m);
  red[threadIdx.x] = s; __syncthreads();
  for (int st = 128; st > 0; st >>= 1) {
    if ((int)threadIdx.x < st) red[threadIdx.x] += red[threadIdx.x + st];
    __syncthreads();
  }
  s = red[0];
  float inv = 1.f / s;
  float* sm = out + (size_t)BB * NCLS;
  for (int c = threadIdx.x; c < NCLS; c += blockDim.x)
    sm[(size_t)b * NCLS + c] = expf(lg[c] - m) * inv;
}

// ---------------------------------------------------------------------------
// Host orchestration
// ---------------------------------------------------------------------------
static inline int cdiv(int a, int b) { return (a + b - 1) / b; }

extern "C" void kernel_launch(void* const* d_in, const int* in_sizes, int n_in,
                              void* d_out, int out_size, void* d_ws, size_t ws_size,
                              hipStream_t stream) {
  (void)in_sizes; (void)n_in; (void)out_size; (void)ws_size;

  const int*   input_ids = (const int*)d_in[0];
  const float* embedding = (const float*)d_in[1];
  const float* norm_w    = (const float*)d_in[2];
  const float* in_proj_w = (const float*)d_in[3];
  const float* conv_w    = (const float*)d_in[4];
  const float* conv_b    = (const float*)d_in[5];
  const float* x_proj_w  = (const float*)d_in[6];
  const float* dt_proj_w = (const float*)d_in[7];
  const float* dt_proj_b = (const float*)d_in[8];
  const float* A_log     = (const float*)d_in[9];
  const float* D_param   = (const float*)d_in[10];
  const float* out_proj_w= (const float*)d_in[11];
  const float* normf_w   = (const float*)d_in[12];
  const float* fc_w      = (const float*)d_in[13];
  const float* fc_b      = (const float*)d_in[14];
  float* out = (float*)d_out;

  // ---- workspace layout ----
  char* base = (char*)d_ws;
  size_t off = 0;
  auto alloc = [&](size_t bytes) {
    size_t o = off; off += (bytes + 255) & ~(size_t)255; return o;
  };
  float*  x_f    = (float*) (base + alloc((size_t)MROWS * DM * 4));
  __bf16* xn_bf  = (__bf16*)(base + alloc((size_t)MROWS * DM * 2));
  __bf16* wT     = (__bf16*)(base + alloc((size_t)(2 * DI) * DM * 2)); // largest W^T, reused
  float*  xr     = (float*) (base + alloc((size_t)MROWS * 2 * DI * 4));
  float*  xc_f   = (float*) (base + alloc((size_t)MROWS * DI * 4));
  __bf16* xc_bf  = (__bf16*)(base + alloc((size_t)MROWS * DI * 2));
  float*  x_dbl  = (float*) (base + alloc((size_t)MROWS * (DR + 2 * NS) * 4));
  __bf16* dlt_bf = (__bf16*)(base + alloc((size_t)MROWS * DR * 2));
  float*  dtbuf  = (float*) (base + alloc((size_t)MROWS * DI * 4));
  float*  yscan  = (float*) (base + alloc((size_t)MROWS * DI * 4));
  __bf16* y_bf   = (__bf16*)(base + alloc((size_t)MROWS * DI * 2));
  float*  tmp    = (float*) (base + alloc((size_t)MROWS * DM * 4));
  float*  xf     = (float*) (base + alloc((size_t)BB * DM * 4));

  const int TB = 256;                          // 8 waves per block
  // exact grids: (M/16)*(N/NPW) waves, all multiples of 8
  constexpr int GB_INPROJ = (MROWS / 16) * ((2 * DI) / 64) / 8;  // 1024
  constexpr int GB_XPROJ  = (MROWS / 16) * (96 / 32) / 8;        // 48
  constexpr int GB_DTPROJ = (MROWS / 16) * (DI / 64) / 8;        // 512
  constexpr int GB_OPROJ  = (MROWS / 16) * (DM / 64) / 8;        // 256

  // ---- embedding ----
  embed_kernel<<<cdiv(MROWS * DM, TB), TB, 0, stream>>>(input_ids, embedding,
                                                        x_f, MROWS * DM);

  for (int l = 0; l < 2; ++l) {
    const float* nw   = norm_w     + (size_t)l * DM;
    const float* ipw  = in_proj_w  + (size_t)l * DM * (2 * DI);
    const float* cw   = conv_w     + (size_t)l * DI * 4;
    const float* cb   = conv_b     + (size_t)l * DI;
    const float* xpw  = x_proj_w   + (size_t)l * DI * (DR + 2 * NS);
    const float* dpw  = dt_proj_w  + (size_t)l * DR * DI;
    const float* dpb  = dt_proj_b  + (size_t)l * DI;
    const float* al   = A_log      + (size_t)l * DI * NS;
    const float* Dp   = D_param    + (size_t)l * DI;
    const float* opw  = out_proj_w + (size_t)l * DI * DM;

    // rmsnorm -> bf16
    rmsnorm_bf16_kernel<<<MROWS, TB, 0, stream>>>(x_f, nw, xn_bf, DM);

    // in_proj: [2048 x 1024] x [1024 x 4096]
    transpose_bf16_kernel<<<cdiv(DM * 2 * DI, TB), TB, 0, stream>>>(ipw, wT, DM, 2 * DI);
    gemm_bf16_wmma_kernel<MROWS, 2 * DI, DM, 64>
        <<<GB_INPROJ, TB, 0, stream>>>(xn_bf, wT, xr);

    // causal conv + bias on xi
    conv_kernel<<<cdiv(MROWS * DI, TB), TB, 0, stream>>>(xr, cw, cb, xc_f, xc_bf,
                                                         MROWS * DI);

    // x_proj: [2048 x 2048] x [2048 x 96]
    transpose_bf16_kernel<<<cdiv(DI * (DR + 2 * NS), TB), TB, 0, stream>>>(
        xpw, wT, DI, DR + 2 * NS);
    gemm_bf16_wmma_kernel<MROWS, DR + 2 * NS, DI, 32>
        <<<GB_XPROJ, TB, 0, stream>>>(xc_bf, wT, x_dbl);

    // dt_proj: [2048 x 64] x [64 x 2048], then softplus(+bias)
    delta_bf16_kernel<<<cdiv(MROWS * DR, TB), TB, 0, stream>>>(x_dbl, dlt_bf,
                                                               MROWS * DR);
    transpose_bf16_kernel<<<cdiv(DR * DI, TB), TB, 0, stream>>>(dpw, wT, DR, DI);
    gemm_bf16_wmma_kernel<MROWS, DI, DR, 64>
        <<<GB_DTPROJ, TB, 0, stream>>>(dlt_bf, wT, dtbuf);
    softplus_kernel<<<cdiv(MROWS * DI, TB), TB, 0, stream>>>(dtbuf, dpb, MROWS * DI);

    // selective scan (sequential over L, states in registers)
    scan_kernel<<<cdiv(BB * DI, TB), TB, 0, stream>>>(dtbuf, xc_f, x_dbl, al, yscan);

    // gating -> bf16
    gate_kernel<<<cdiv(MROWS * DI, TB), TB, 0, stream>>>(yscan, xc_f, Dp, xr, y_bf,
                                                         MROWS * DI);

    // out_proj: [2048 x 2048] x [2048 x 1024], residual add
    transpose_bf16_kernel<<<cdiv(DI * DM, TB), TB, 0, stream>>>(opw, wT, DI, DM);
    gemm_bf16_wmma_kernel<MROWS, DM, DI, 64>
        <<<GB_OPROJ, TB, 0, stream>>>(y_bf, wT, tmp);
    add_kernel<<<cdiv(MROWS * DM, TB), TB, 0, stream>>>(x_f, tmp, MROWS * DM);
  }

  // final norm (last token only), fc head, softmax
  rmsnorm_last_kernel<<<BB, TB, 0, stream>>>(x_f, normf_w, xf);
  fc_kernel<<<cdiv(BB * NCLS, TB), TB, 0, stream>>>(xf, fc_w, fc_b, out, BB * NCLS);
  softmax_kernel<<<BB, TB, 0, stream>>>(out);
}
